// _TemplateTriMul_20873541058561
// MI455X (gfx1250) — matrix operand order repaired and checked by reading, not compiled, not run on hardware
//
#include <hip/hip_runtime.h>

// ---------------------------------------------------------------------------
// TriMul (outgoing) for MI455X / gfx1250, wave32 + v_wmma_f32_16x16x32_bf16.
// Memory-bound (~58 GFLOP vs <1 GB HBM traffic @ 23.3 TB/s): bf16 operands,
// fp32 WMMA accumulation. a_t/b_t (67 MB each, bf16) both fit in 192 MB L2,
// so the 128 batched NT-GEMMs of the einsum stream from L2.
// Round-2 fixes: native v_rcp/v_rsq for sigmoid/LN (kill IEEE div chains),
// batched load clauses before WMMA groups in k3, LDS repack of k3 output so
// global stores are 32B-contiguous instead of 4B-scattered.
// ---------------------------------------------------------------------------

typedef __bf16 bf16_t;
typedef __attribute__((ext_vector_type(16))) __bf16 v16bf;
typedef __attribute__((ext_vector_type(8)))  __bf16 v8bf;
typedef __attribute__((ext_vector_type(8)))  float   v8f;
typedef __attribute__((ext_vector_type(4)))  float   v4f;

#define NDIM 512
#define DDIM 64
#define HDIM 128
#define POSN (NDIM * NDIM)   // 262144 (i,k) positions

__device__ __forceinline__ bf16_t f2bf(float f) {
  unsigned u = __builtin_bit_cast(unsigned, f);
  u += 0x7FFFu + ((u >> 16) & 1u);               // round-to-nearest-even
  unsigned short s = (unsigned short)(u >> 16);
  return __builtin_bit_cast(bf16_t, s);
}
__device__ __forceinline__ float bf2f(bf16_t b) {
  unsigned short s = __builtin_bit_cast(unsigned short, b);
  unsigned u = ((unsigned)s) << 16;
  return __builtin_bit_cast(float, u);
}
// native v_rcp_f32 / v_rsq_f32 (~1 ulp) instead of IEEE div/sqrt expansions
__device__ __forceinline__ float sigm(float x) {
  return __builtin_amdgcn_rcpf(1.0f + __expf(-x));
}
__device__ __forceinline__ float rsq(float x) {
  return __builtin_amdgcn_rsqf(x);
}

// --- WMMA fragment builders (ISA 7.12.2 VGPR layouts, wave32) --------------
// A (16x32 bf16, MxK): lane L holds row m=L&15; hi=L>>4.
//   elements 0..7  -> k = k0 + 8*hi + e
//   elements 8..15 -> k = k0 + 16 + 8*hi + (e-8)
__device__ __forceinline__ v16bf load_afrag(const bf16_t* rowp, int k0, int hi) {
  union { v16bf v; v8bf h[2]; } u;
  u.h[0] = *(const v8bf*)(rowp + k0 + 8 * hi);
  u.h[1] = *(const v8bf*)(rowp + k0 + 16 + 8 * hi);
  return u.v;
}
// B (32x16 bf16, KxN): lane L holds col n=L&15; same k order as A fragment.
// rowp = pointer to row n of B^T (unit-stride over k).
__device__ __forceinline__ v16bf load_bfrag(const bf16_t* rowp, int k0, int hi) {
  union { v16bf v; v8bf h[2]; } u;
  u.h[0] = *(const v8bf*)(rowp + k0 + 8 * hi);
  u.h[1] = *(const v8bf*)(rowp + k0 + 16 + 8 * hi);
  return u.v;
}
__device__ __forceinline__ v8f wmma_bf16(v16bf a, v16bf b, v8f c) {
  return __builtin_amdgcn_wmma_f32_16x16x32_bf16(false, a, false, b,
                                                 (short)0, c, false, false);
}

// --- k0: convert + transpose weights to bf16 (B^T layout: [n][k]) ----------
__global__ void k0_weights(const float* __restrict__ Wp, const float* __restrict__ Wg,
                           const float* __restrict__ Wout, const float* __restrict__ Wgate,
                           bf16_t* __restrict__ wp_t, bf16_t* __restrict__ wg_t,
                           bf16_t* __restrict__ wout_t, bf16_t* __restrict__ wgate_t) {
  int t = blockIdx.x * blockDim.x + threadIdx.x;
  if (t < 64 * 256) { int k = t >> 8, n = t & 255; wp_t[n * 64 + k]  = f2bf(Wp[t]); }
  if (t < 64 * 256) { int k = t >> 8, n = t & 255; wg_t[n * 64 + k]  = f2bf(Wg[t]); }
  if (t < 128 * 64) { int k = t >> 6, n = t & 63;  wout_t[n * 128 + k] = f2bf(Wout[t]); }
  if (t < 64 * 64)  { int k = t >> 6, n = t & 63;  wgate_t[n * 64 + k] = f2bf(Wgate[t]); }
}

// --- k1: LN over d=64, wave per row, bf16 out ------------------------------
__global__ void __launch_bounds__(256) k1_ln_in(const float* __restrict__ z,
                                                const float* __restrict__ w,
                                                const float* __restrict__ b,
                                                bf16_t* __restrict__ h) {
  int wave = (blockIdx.x * blockDim.x + threadIdx.x) >> 5;
  int lane = threadIdx.x & 31;
  const float* row = z + (size_t)wave * DDIM;
  float x0 = row[lane], x1 = row[lane + 32];
  float s = x0 + x1, ss = x0 * x0 + x1 * x1;
  #pragma unroll
  for (int m = 16; m >= 1; m >>= 1) { s += __shfl_xor(s, m); ss += __shfl_xor(ss, m); }
  float mu  = s * (1.0f / 64.0f);
  float var = ss * (1.0f / 64.0f) - mu * mu;
  float inv = rsq(var + 1e-5f);
  bf16_t* orow = h + (size_t)wave * DDIM;
  orow[lane]      = f2bf((x0 - mu) * inv * w[lane]      + b[lane]);
  orow[lane + 32] = f2bf((x1 - mu) * inv * w[lane + 32] + b[lane + 32]);
}

// --- k2: projections via WMMA; fused sigmoid/product; channel-major a,b ----
__global__ void __launch_bounds__(256) k2_proj(const bf16_t* __restrict__ h,
                                               const bf16_t* __restrict__ wp_t,
                                               const bf16_t* __restrict__ wg_t,
                                               const bf16_t* __restrict__ wgate_t,
                                               bf16_t* __restrict__ a_t,
                                               bf16_t* __restrict__ b_t,
                                               bf16_t* __restrict__ gate_t) {
  int wid  = (blockIdx.x * blockDim.x + threadIdx.x) >> 5;  // 16-row tile id
  int lane = threadIdx.x & 31;
  int m = lane & 15, hi = lane >> 4;
  size_t pos0 = (size_t)wid * 16;
  const bf16_t* arow = h + (pos0 + (size_t)m) * DDIM;
  v16bf a0 = load_afrag(arow, 0,  hi);
  v16bf a1 = load_afrag(arow, 32, hi);
  size_t wr = pos0 + (size_t)(8 * hi);   // 8 consecutive positions per lane

  // p = h@Wp, g = sigmoid(h@Wg); pg split -> a (cols 0..127) / b (128..255)
  for (int nt = 0; nt < 16; ++nt) {
    const bf16_t* bp = wp_t + (size_t)(nt * 16 + m) * 64;
    const bf16_t* bg = wg_t + (size_t)(nt * 16 + m) * 64;
    // batch all four B-fragment loads before the WMMA group
    v16bf bp0 = load_bfrag(bp, 0,  hi);
    v16bf bp1 = load_bfrag(bp, 32, hi);
    v16bf bg0 = load_bfrag(bg, 0,  hi);
    v16bf bg1 = load_bfrag(bg, 32, hi);
    v8f accp = {}, accg = {};
    accp = wmma_bf16(a0, bp0, accp);
    accp = wmma_bf16(a1, bp1, accp);
    accg = wmma_bf16(a0, bg0, accg);
    accg = wmma_bf16(a1, bg1, accg);
    v8bf outv;
    #pragma unroll
    for (int v = 0; v < 8; ++v) outv[v] = f2bf(accp[v] * sigm(accg[v]));
    int col = nt * 16 + m;
    bf16_t* dst = (col < HDIM) ? (a_t + (size_t)col * POSN + wr)
                               : (b_t + (size_t)(col - HDIM) * POSN + wr);
    *(v8bf*)dst = outv;
  }
  // gate = sigmoid(h@Wgate), stored channel-major
  for (int nt = 0; nt < 4; ++nt) {
    const bf16_t* bgt = wgate_t + (size_t)(nt * 16 + m) * 64;
    v16bf bt0 = load_bfrag(bgt, 0,  hi);
    v16bf bt1 = load_bfrag(bgt, 32, hi);
    v8f acc = {};
    acc = wmma_bf16(a0, bt0, acc);
    acc = wmma_bf16(a1, bt1, acc);
    v8bf outv;
    #pragma unroll
    for (int v = 0; v < 8; ++v) outv[v] = f2bf(sigm(acc[v]));
    int col = nt * 16 + m;
    *(v8bf*)(gate_t + (size_t)col * POSN + wr) = outv;
  }
}

// --- k3: out[i,j,c] = sum_k a[i,k,c]*b[j,k,c]  (128 NT-GEMMs, L2-resident) -
// Block: 8 waves = 8 channels (c0..c0+7); each wave: 16(i) x 64(j), K=512.
// All loads of a k-step are issued as one clause before the 4 WMMAs; the
// epilogue repacks through LDS so out_e stores are 32B-contiguous per (i,j).
__global__ void __launch_bounds__(256) k3_einsum(const bf16_t* __restrict__ a_t,
                                                 const bf16_t* __restrict__ b_t,
                                                 float* __restrict__ out_e) {
  __shared__ __align__(16) float tile[16][64][8];   // 32 KB: [i][j][c-in-block]
  int w    = threadIdx.x >> 5;
  int lane = threadIdx.x & 31;
  int m = lane & 15, hi = lane >> 4;
  int c0 = blockIdx.z * 8;
  int c  = c0 + w;
  int i0 = blockIdx.y * 16;
  int j0 = blockIdx.x * 64;
  const bf16_t* abase = a_t + (size_t)c * POSN + (size_t)(i0 + m) * NDIM;
  const bf16_t* bbase = b_t + (size_t)c * POSN + (size_t)(j0 + m) * NDIM;
  v8f acc0 = {}, acc1 = {}, acc2 = {}, acc3 = {};
  #pragma unroll 2
  for (int k0 = 0; k0 < NDIM; k0 += 32) {
    v16bf af  = load_afrag(abase, k0, hi);
    v16bf bf0 = load_bfrag(bbase,             k0, hi);
    v16bf bf1 = load_bfrag(bbase + 16 * NDIM, k0, hi);
    v16bf bf2 = load_bfrag(bbase + 32 * NDIM, k0, hi);
    v16bf bf3 = load_bfrag(bbase + 48 * NDIM, k0, hi);
    acc0 = wmma_bf16(af, bf0, acc0);
    acc1 = wmma_bf16(af, bf1, acc1);
    acc2 = wmma_bf16(af, bf2, acc2);
    acc3 = wmma_bf16(af, bf3, acc3);
  }
  // D layout: lane holds col n=lane&15 of its j-tile; VGPR v holds row v+8*hi.
  #pragma unroll
  for (int v = 0; v < 8; ++v) {
    int i = v + 8 * hi;
    tile[i][ 0 + m][w] = acc0[v];
    tile[i][16 + m][w] = acc1[v];
    tile[i][32 + m][w] = acc2[v];
    tile[i][48 + m][w] = acc3[v];
  }
  __syncthreads();
  // cooperative 32B-contiguous stores: 1024 (i,j) positions / 256 threads
  for (int t = threadIdx.x; t < 16 * 64; t += 256) {
    int i = t >> 6, j = t & 63;
    const float* src = &tile[i][j][0];
    v4f lo = *(const v4f*)src;
    v4f hi4 = *(const v4f*)(src + 4);
    float* dst = out_e + ((size_t)(i0 + i) * NDIM + (size_t)(j0 + j)) * HDIM + c0;
    *(v4f*)dst       = lo;
    *(v4f*)(dst + 4) = hi4;
  }
}

// --- k4: LN over hidden=128, @Wout via WMMA, * gate, fp32 out --------------
__global__ void __launch_bounds__(256) k4_out(const float* __restrict__ out_e,
                                              const float* __restrict__ lw,
                                              const float* __restrict__ lb,
                                              const bf16_t* __restrict__ wout_t,
                                              const bf16_t* __restrict__ gate_t,
                                              float* __restrict__ out) {
  __shared__ __align__(16) bf16_t h2[8][16][HDIM];   // 32 KB of the 320 KB LDS
  int w    = threadIdx.x >> 5;
  int lane = threadIdx.x & 31;
  int m = lane & 15, hi = lane >> 4;
  size_t pos0 = ((size_t)blockIdx.x * 8 + (size_t)w) * 16;

  for (int r = 0; r < 16; ++r) {
    const float* row = out_e + (pos0 + r) * HDIM;
    v4f x = *(const v4f*)(row + lane * 4);
    float s  = x[0] + x[1] + x[2] + x[3];
    float ss = x[0]*x[0] + x[1]*x[1] + x[2]*x[2] + x[3]*x[3];
    #pragma unroll
    for (int mk = 16; mk >= 1; mk >>= 1) { s += __shfl_xor(s, mk); ss += __shfl_xor(ss, mk); }
    float mu  = s * (1.0f / 128.0f);
    float var = ss * (1.0f / 128.0f) - mu * mu;
    float inv = rsq(var + 1e-5f);
    #pragma unroll
    for (int e = 0; e < 4; ++e) {
      int col = lane * 4 + e;
      h2[w][r][col] = f2bf((x[e] - mu) * inv * lw[col] + lb[col]);
    }
  }
  __syncthreads();

  const bf16_t* arow = h2[w][m];
  v16bf af[4];
  #pragma unroll
  for (int ks = 0; ks < 4; ++ks) af[ks] = load_afrag(arow, ks * 32, hi);

  #pragma unroll
  for (int nt = 0; nt < 4; ++nt) {
    const bf16_t* brow = wout_t + (size_t)(nt * 16 + m) * HDIM;
    v16bf b0 = load_bfrag(brow,  0, hi);
    v16bf b1 = load_bfrag(brow, 32, hi);
    v16bf b2 = load_bfrag(brow, 64, hi);
    v16bf b3 = load_bfrag(brow, 96, hi);
    v8f acc = {};
    acc = wmma_bf16(af[0], b0, acc);
    acc = wmma_bf16(af[1], b1, acc);
    acc = wmma_bf16(af[2], b2, acc);
    acc = wmma_bf16(af[3], b3, acc);
    int col = nt * 16 + m;
    const bf16_t* g = gate_t + (size_t)col * POSN + pos0 + (size_t)(8 * hi);
    #pragma unroll
    for (int v = 0; v < 8; ++v)
      out[(pos0 + (size_t)(v + 8 * hi)) * DDIM + col] = acc[v] * bf2f(g[v]);
  }
}

// ---------------------------------------------------------------------------
extern "C" void kernel_launch(void* const* d_in, const int* in_sizes, int n_in,
                              void* d_out, int out_size, void* d_ws, size_t ws_size,
                              hipStream_t stream) {
  (void)in_sizes; (void)n_in; (void)out_size; (void)ws_size;
  const float* z        = (const float*)d_in[0];
  const float* ln_in_w  = (const float*)d_in[1];
  const float* ln_in_b  = (const float*)d_in[2];
  const float* Wp       = (const float*)d_in[3];
  const float* Wg       = (const float*)d_in[4];
  const float* ln_out_w = (const float*)d_in[5];
  const float* ln_out_b = (const float*)d_in[6];
  const float* Wout     = (const float*)d_in[7];
  const float* Wgate    = (const float*)d_in[8];
  float* out = (float*)d_out;

  char* ws = (char*)d_ws;
  size_t off = 0;
  auto take = [&](size_t bytes) -> char* {
    char* p = ws + off;
    off += (bytes + 255) & ~(size_t)255;
    return p;
  };
  bf16_t* h       = (bf16_t*)take((size_t)POSN * DDIM * 2);        //  33.5 MB
  bf16_t* a_t     = (bf16_t*)take((size_t)HDIM * POSN * 2);        //  67.1 MB
  bf16_t* b_t     = (bf16_t*)take((size_t)HDIM * POSN * 2);        //  67.1 MB
  bf16_t* gate_t  = (bf16_t*)take((size_t)DDIM * POSN * 2);        //  33.5 MB
  float*  out_e   = (float*) take((size_t)POSN * HDIM * 4);        // 134.2 MB
  bf16_t* wp_t    = (bf16_t*)take(256 * 64 * 2);
  bf16_t* wg_t    = (bf16_t*)take(256 * 64 * 2);
  bf16_t* wgate_t = (bf16_t*)take(64 * 64 * 2);
  bf16_t* wout_t  = (bf16_t*)take(64 * 128 * 2);

  k0_weights<<<64, 256, 0, stream>>>(Wp, Wg, Wout, Wgate, wp_t, wg_t, wout_t, wgate_t);
  k1_ln_in<<<POSN / 8, 256, 0, stream>>>(z, ln_in_w, ln_in_b, h);
  k2_proj<<<POSN / 16 / 8, 256, 0, stream>>>(h, wp_t, wg_t, wgate_t, a_t, b_t, gate_t);
  k3_einsum<<<dim3(8, 32, 16), 256, 0, stream>>>(a_t, b_t, out_e);
  k4_out<<<POSN / 16 / 8, 256, 0, stream>>>(out_e, ln_out_w, ln_out_b, wout_t, gate_t, out);
}
